// PolicyNetV2Torch_28183575396407
// MI455X (gfx1250) — matrix operand order repaired and checked by reading, not compiled, not run on hardware
//
#include <hip/hip_runtime.h>
#include <hip/hip_bf16.h>

// ---------------------------------------------------------------------------
// PolicyNet fused forward for gfx1250 (MI455X), wave32 + WMMA bf16.
//   Kernel 1: one workgroup per batch item  -> entity embed, 2 attention
//             blocks (LN + QKV/O projections + softmax attention) fully in
//             LDS, masked max-pool, ego embed; writes trunk[B,128] (bf16) to ws.
//   Kernel 2: MLP head fc1(128->256)+fc2(256->256) as WMMA GEMMs over
//             32-row batch tiles; actor/value heads as VALU dots.
// All matmuls use v_wmma_f32_16x16x32_bf16 (fp32 accumulate).
// B-operand sources are kept K-major (transposed) in LDS ; B operands are
// gathered per element at a k-row stride, A from 16-byte row loads.
// Pure (no-convert) global->LDS staging uses GLOBAL_LOAD_ASYNC_TO_LDS_B128
// (ASYNCcnt) instead of load+store through VGPRs.
// ---------------------------------------------------------------------------

#define NB      4096
#define AST     72           // padded bf16 row stride for [64 x 64] tiles

typedef __attribute__((ext_vector_type(16))) __bf16 v16bf;
typedef __attribute__((ext_vector_type(8)))  __bf16 v8bf;
typedef __attribute__((ext_vector_type(8)))  float  v8f;

__device__ __forceinline__ v8bf ldv8(const __bf16* p) { return *(const v8bf*)p; }

__device__ __forceinline__ v16bf cat8(v8bf lo, v8bf hi) {
    return __builtin_shufflevector(lo, hi, 0, 1, 2, 3, 4, 5, 6, 7,
                                           8, 9, 10, 11, 12, 13, 14, 15);
}

// A operand (16x32 MxK, 16-bit): lane holds row M=lane&15; element j holds
// K = khalf+j (j<8) / 16+khalf+j-8 (j>=8), khalf = 8*(lane>=16).
__device__ __forceinline__ v16bf load_a(const __bf16* arow, int k0, int lane) {
    int khalf = (lane >> 4) << 3;
    return cat8(ldv8(arow + k0 + khalf), ldv8(arow + k0 + 16 + khalf));
}

// B operand (32x16 KxN, 16-bit): lane holds column N=lane&15; element j holds
// K = khalf+j (j<8) / 16+khalf+j-8 (j>=8), khalf = 8*(lane>=16); strided gather.
__device__ __forceinline__ v16bf load_b(const __bf16* col, int ld, int lane) {
    int khalf = (lane >> 4) << 3;
    v16bf r;
#pragma unroll
    for (int j = 0; j < 16; ++j) r[j] = col[(j < 8 ? khalf + j : 8 + khalf + j) * ld];
    return r;
}

__device__ __forceinline__ v8f wmma_bf16(v16bf a, v16bf b, v8f c) {
    return __builtin_amdgcn_wmma_f32_16x16x32_bf16(
        false, a, false, b, (short)0, c, false, false);
}

// Async DMA: 16 bytes global -> LDS per active lane (ASYNCcnt tracked).
// Generic LDS pointer truncated to 32 bits == LDS offset (ISA: addr[31:0]).
__device__ __forceinline__ void async_copy_b128(void* lds_dst, const void* gsrc) {
    asm volatile("global_load_async_to_lds_b128 %0, %1, off"
                 :: "v"((unsigned)(unsigned long long)lds_dst),
                    "v"((unsigned long long)gsrc)
                 : "memory");
}
__device__ __forceinline__ void wait_async0() {
    asm volatile("s_wait_asynccnt 0x0" ::: "memory");
}

// Stage a [64,64] fp32 weight (row-major w[out][in]) into LDS K-major bf16:
// wtT[k*64 + n] = w[n][k].  256 threads cover 64 k-rows x 4 n-chunks.
__device__ __forceinline__ void stage_w64T(__bf16* wtT, const float* __restrict__ w, int tid) {
    int k = tid & 63, n0 = (tid >> 6) << 4;
    v16bf pk;
#pragma unroll
    for (int j = 0; j < 16; ++j) pk[j] = (__bf16)w[(n0 + j) * 64 + k];
    *(v16bf*)&wtT[k * 64 + n0] = pk;         // 32B aligned -> 2x ds_store_b128
}

// Y[64,64] = A[64,64] @ W^T (+bias).  A bf16 row-major stride AST,
// W staged K-major [64][64].  dstT: K-major bf16 (packed b128 stores);
// dstB: row-major bf16; dstF: fp32 residual add.
__device__ __forceinline__ void gemm64(const __bf16* __restrict__ A,
                                       const __bf16* __restrict__ W,
                                       const float*  __restrict__ bias,
                                       __bf16* dstB, __bf16* dstT, float* dstF,
                                       int lane, int wave) {
    for (int t = 0; t < 2; ++t) {
        int tile = wave + t * 8;                  // 16 tiles over 8 waves
        int m0 = (tile >> 2) * 16, n0 = (tile & 3) * 16;
        const __bf16* arow = A + (m0 + (lane & 15)) * AST;
        v8f c = {};
        for (int k0 = 0; k0 < 64; k0 += 32) {
            v16bf a  = load_a(arow, k0, lane);
            v16bf bm = load_b(&W[k0 * 64 + n0 + (lane & 15)], 64, lane);
            c = wmma_bf16(a, bm, c);
        }
        int n = n0 + (lane & 15);
        float bv = bias[n];
        if (dstT) {                               // transposed: contiguous in M
            v8bf pk;
#pragma unroll
            for (int v = 0; v < 8; ++v) pk[v] = (__bf16)(c[v] + bv);
            *(v8bf*)&dstT[n * AST + m0 + ((lane >> 4) << 3)] = pk;
        } else if (dstB) {
#pragma unroll
            for (int v = 0; v < 8; ++v) {
                int s = m0 + v + ((lane >> 4) << 3);
                dstB[s * AST + n] = (__bf16)(c[v] + bv);
            }
        } else {
#pragma unroll
            for (int v = 0; v < 8; ++v) {
                int s = m0 + v + ((lane >> 4) << 3);
                dstF[s * 64 + n] += c[v] + bv;    // residual: += (out + bias)
            }
        }
    }
}

__device__ void attn_block(int tid, int lane, int wave, int n_ent,
                           float* xf, __bf16* uA, __bf16* uB,
                           __bf16* qb, __bf16* kT, __bf16* vb,
                           const float* __restrict__ lng, const float* __restrict__ lnb,
                           const float* __restrict__ wq, const float* __restrict__ bq,
                           const float* __restrict__ wk, const float* __restrict__ bk,
                           const float* __restrict__ wv, const float* __restrict__ bv,
                           const float* __restrict__ wo, const float* __restrict__ bo) {
    __builtin_prefetch(wq, 0, 1);                 // global_prefetch_b8 (L2)
    __builtin_prefetch(wk, 0, 1);
    __builtin_prefetch(wv, 0, 1);
    __builtin_prefetch(wo, 0, 1);

    // ---- LayerNorm (fp32) -> uA (bf16), one row per thread (waves 0-1) ----
    if (tid < 64) {
        int s = tid;
        float mu = 0.f;
        for (int e = 0; e < 64; ++e) mu += xf[s * 64 + e];
        mu *= (1.f / 64.f);
        float var = 0.f;
        for (int e = 0; e < 64; ++e) { float d = xf[s * 64 + e] - mu; var += d * d; }
        var *= (1.f / 64.f);
        float rinv = rsqrtf(var + 1e-4f);
        for (int e0 = 0; e0 < 64; e0 += 8) {
            v8bf pk;
#pragma unroll
            for (int j = 0; j < 8; ++j)
                pk[j] = (__bf16)((xf[s * 64 + e0 + j] - mu) * rinv * lng[e0 + j] + lnb[e0 + j]);
            *(v8bf*)&uA[s * AST + e0] = pk;
        }
    }
    __syncthreads();

    // ---- Q (row-major), K (K-major/transposed), V (row-major) ----
    stage_w64T(uB, wq, tid); __syncthreads();
    gemm64(uA, uB, bq, qb, nullptr, nullptr, lane, wave); __syncthreads();
    stage_w64T(uB, wk, tid); __syncthreads();
    gemm64(uA, uB, bk, nullptr, kT, nullptr, lane, wave); __syncthreads();
    stage_w64T(uB, wv, tid); __syncthreads();
    gemm64(uA, uB, bv, vb, nullptr, nullptr, lane, wave); __syncthreads();

    // ---- heads (sequential; scores reuse uB) ----
    __bf16* sc = uB;
    for (int h = 0; h < 4; ++h) {
        int h16 = h * 16;
        // scores = Q K^T * 0.25, clip, mask  (K-dim = 16, zero-padded to 32)
        for (int t = 0; t < 2; ++t) {
            int tile = wave + t * 8;
            int m0 = (tile >> 2) * 16, n0 = (tile & 3) * 16;
            int khalf = (lane >> 4) << 3;
            v8bf z8 = {};
            v16bf a = cat8(ldv8(&qb[(m0 + (lane & 15)) * AST + h16 + khalf]), z8);
            v16bf bm = {};
#pragma unroll
            for (int j = 0; j < 8; ++j) bm[j] = kT[(h16 + khalf + j) * AST + n0 + (lane & 15)];
            v8f c = {};
            c = wmma_bf16(a, bm, c);
            int tcol = n0 + (lane & 15);
#pragma unroll
            for (int v = 0; v < 8; ++v) {
                int s = m0 + v + ((lane >> 4) << 3);
                float val = c[v] * 0.25f;
                val = fminf(fmaxf(val, -20.f), 20.f);
                if (tcol >= n_ent) val = -10000.f;
                sc[s * AST + tcol] = (__bf16)val;
            }
        }
        __syncthreads();
        // softmax per row (64 rows -> waves 0-1)
        if (tid < 64) {
            int s = tid;
            float m = -1e30f;
            for (int t = 0; t < 64; ++t) m = fmaxf(m, (float)sc[s * AST + t]);
            float ssum = 0.f;
            for (int t = 0; t < 64; ++t) {
                float x = __expf((float)sc[s * AST + t] - m);
                ssum += x;
                sc[s * AST + t] = (__bf16)x;
            }
            float inv = 1.f / ssum;
            for (int t = 0; t < 64; ++t)
                sc[s * AST + t] = (__bf16)((float)sc[s * AST + t] * inv);
        }
        __syncthreads();
        // out_h = attn @ V  -> uA[:, h*16 : h*16+16]   (4 m-tiles, waves 0-3)
        if (wave < 4) {
            int m0 = wave * 16;
            const __bf16* arow = sc + (m0 + (lane & 15)) * AST;
            v8f c = {};
            for (int k0 = 0; k0 < 64; k0 += 32) {
                v16bf a  = load_a(arow, k0, lane);
                v16bf bm = load_b(&vb[k0 * AST + h16 + (lane & 15)], AST, lane);
                c = wmma_bf16(a, bm, c);
            }
            int d = lane & 15;
#pragma unroll
            for (int v = 0; v < 8; ++v) {
                int s = m0 + v + ((lane >> 4) << 3);
                uA[s * AST + h16 + d] = (__bf16)c[v];
            }
        }
        __syncthreads();
    }

    // ---- output projection + residual into xf ----
    stage_w64T(uB, wo, tid); __syncthreads();
    gemm64(uA, uB, bo, nullptr, nullptr, xf, lane, wave); __syncthreads();
}

__global__ void __launch_bounds__(256)
policy_attn_kernel(const float* __restrict__ ego, const float* __restrict__ entities,
                   const int* __restrict__ n_entities,
                   const float* __restrict__ ego_w, const float* __restrict__ ego_b,
                   const float* __restrict__ ent_w, const float* __restrict__ ent_b,
                   const float* __restrict__ ln1g, const float* __restrict__ ln1b,
                   const float* __restrict__ wq1, const float* __restrict__ bq1,
                   const float* __restrict__ wk1, const float* __restrict__ bk1,
                   const float* __restrict__ wv1, const float* __restrict__ bv1,
                   const float* __restrict__ wo1, const float* __restrict__ bo1,
                   const float* __restrict__ ln2g, const float* __restrict__ ln2b,
                   const float* __restrict__ wq2, const float* __restrict__ bq2,
                   const float* __restrict__ wk2, const float* __restrict__ bk2,
                   const float* __restrict__ wv2, const float* __restrict__ bv2,
                   const float* __restrict__ wo2, const float* __restrict__ bo2,
                   __bf16* __restrict__ trunk) {
    __shared__ __align__(32) float  xf[64 * 64];      // 16 KB fp32 activations
    __shared__ __align__(32) __bf16 uA[64 * AST];     // LN out / attn out
    __shared__ __align__(32) __bf16 uB[64 * AST];     // weights / scores / ent stage
    __shared__ __align__(32) __bf16 qb[64 * AST];     // Q row-major
    __shared__ __align__(32) __bf16 kT[64 * AST];     // K transposed [e][s]
    __shared__ __align__(32) __bf16 vb[64 * AST];     // V row-major

    int tid = threadIdx.x, lane = tid & 31, wave = tid >> 5;
    int b = blockIdx.x;
    int n_ent = n_entities[b];

    // ---- entity rows: async DMA global -> LDS (no conversion, ASYNCcnt) ----
    float* entf = (float*)uB;                         // 64*12 floats fit in uB
    if (tid < 192)                                    // 192 x 16B = 3072 B
        async_copy_b128(entf + tid * 4, entities + b * (64 * 12) + tid * 4);
    wait_async0();
    __syncthreads();

    // ---- entity embed: xf = relu(entities @ ent_w^T + ent_b), K=12 (VALU) ----
    for (int i = tid; i < 64 * 64; i += 256) {
        int s = i >> 6, e = i & 63;
        float acc = ent_b[e];
#pragma unroll
        for (int k = 0; k < 12; ++k) acc += entf[s * 12 + k] * ent_w[e * 12 + k];
        xf[i] = fmaxf(acc, 0.f);
    }
    // ---- ego embed -> trunk[:, 0:64] (independent of attention path) ----
    if (tid < 64) {
        int e = tid;
        float acc = ego_b[e];
#pragma unroll
        for (int k = 0; k < 16; ++k) acc += ego[b * 16 + k] * ego_w[e * 16 + k];
        trunk[b * 128 + e] = (__bf16)fmaxf(acc, 0.f);
    }
    __syncthreads();

    attn_block(tid, lane, wave, n_ent, xf, uA, uB, qb, kT, vb,
               ln1g, ln1b, wq1, bq1, wk1, bk1, wv1, bv1, wo1, bo1);
    attn_block(tid, lane, wave, n_ent, xf, uA, uB, qb, kT, vb,
               ln2g, ln2b, wq2, bq2, wk2, bk2, wv2, bv2, wo2, bo2);

    // ---- masked max-pool over entities -> trunk[:, 64:128] ----
    if (tid < 64) {
        int e = tid;
        float p = 0.f;
        if (n_ent > 0) {
            p = -10000.f;
            for (int s = 0; s < n_ent; ++s) p = fmaxf(p, xf[s * 64 + e]);
        }
        trunk[b * 128 + 64 + e] = (__bf16)p;
    }
}

// ---------------------------------------------------------------------------
// MLP head: 32 batch rows per workgroup, fc1/fc2 via WMMA, heads via VALU.
// Trunk (bf16) staged via async DMA; fc weights gathered from global fp32
// (contiguous in k per lane, L2-resident: 384 KB << 192 MB L2).
// ---------------------------------------------------------------------------
__global__ void __launch_bounds__(256)
mlp_head_kernel(const __bf16* __restrict__ trunk,
                const float* __restrict__ fc1_w, const float* __restrict__ fc1_b,
                const float* __restrict__ fc2_w, const float* __restrict__ fc2_b,
                const float* __restrict__ actor_w, const float* __restrict__ actor_b,
                const float* __restrict__ val_w, const float* __restrict__ val_b,
                float* __restrict__ out) {
    __shared__ __align__(32) __bf16 bufA[32 * 256];   // trunk (cols 0..127) then h2
    __shared__ __align__(32) __bf16 bufB[32 * 256];   // h1
    int tid = threadIdx.x, lane = tid & 31, wave = tid >> 5;
    int r0 = blockIdx.x * 32;

    __builtin_prefetch(fc1_w, 0, 1);
    __builtin_prefetch(fc2_w, 0, 1);

    // 32 rows x 128 bf16 = 512 x 16B chunks, async DMA into stride-256 layout
#pragma unroll
    for (int it = 0; it < 2; ++it) {
        int i = tid + it * 256;                       // chunk id
        int r = i >> 4, c0 = (i & 15) * 8;            // 16 chunks per 128-col row
        async_copy_b128(&bufA[r * 256 + c0], &trunk[(r0 + r) * 128 + c0]);
    }
    wait_async0();
    __syncthreads();

    // fc1: [32,128] @ [256,128]^T -> relu -> bufB [32,256]
    for (int tt = 0; tt < 4; ++tt) {
        int tile = wave + tt * 8;                     // 2m x 16n tiles
        int m0 = (tile >> 4) * 16, n0 = (tile & 15) * 16;
        const __bf16* arow = bufA + (m0 + (lane & 15)) * 256;
        v8f c = {};
        for (int k0 = 0; k0 < 128; k0 += 32) {
            v16bf a = load_a(arow, k0, lane);
            v16bf bm;
#pragma unroll
            for (int j = 0; j < 16; ++j)
                bm[j] = (__bf16)fc1_w[(n0 + (lane & 15)) * 128 + k0 + (j < 8 ? ((lane >> 4) << 3) + j : 8 + ((lane >> 4) << 3) + j)];
            c = wmma_bf16(a, bm, c);
        }
        int n = n0 + (lane & 15);
        float bv = fc1_b[n];
#pragma unroll
        for (int v = 0; v < 8; ++v) {
            int r = m0 + v + ((lane >> 4) << 3);
            bufB[r * 256 + n] = (__bf16)fmaxf(c[v] + bv, 0.f);
        }
    }
    __syncthreads();

    // fc2: [32,256] @ [256,256]^T -> relu -> bufA [32,256]
    for (int tt = 0; tt < 4; ++tt) {
        int tile = wave + tt * 8;
        int m0 = (tile >> 4) * 16, n0 = (tile & 15) * 16;
        const __bf16* arow = bufB + (m0 + (lane & 15)) * 256;
        v8f c = {};
        for (int k0 = 0; k0 < 256; k0 += 32) {
            v16bf a = load_a(arow, k0, lane);
            v16bf bm;
#pragma unroll
            for (int j = 0; j < 16; ++j)
                bm[j] = (__bf16)fc2_w[(n0 + (lane & 15)) * 256 + k0 + (j < 8 ? ((lane >> 4) << 3) + j : 8 + ((lane >> 4) << 3) + j)];
            c = wmma_bf16(a, bm, c);
        }
        int n = n0 + (lane & 15);
        float bv = fc2_b[n];
#pragma unroll
        for (int v = 0; v < 8; ++v) {
            int r = m0 + v + ((lane >> 4) << 3);
            bufA[r * 256 + n] = (__bf16)fmaxf(c[v] + bv, 0.f);
        }
    }
    __syncthreads();

    // actor (4) + value (1) heads: one (row, out) dot per thread
    if (tid < 32 * 5) {
        int r = tid / 5, o = tid % 5;
        const float* w = (o < 4) ? (actor_w + o * 256) : val_w;
        float acc = (o < 4) ? actor_b[o] : val_b[0];
        for (int k = 0; k < 256; ++k) acc += (float)bufA[r * 256 + k] * w[k];
        if (o < 4) out[(r0 + r) * 4 + o] = acc;
        else       out[NB * 4 + (r0 + r)] = acc;
    }
}

extern "C" void kernel_launch(void* const* d_in, const int* in_sizes, int n_in,
                              void* d_out, int out_size, void* d_ws, size_t ws_size,
                              hipStream_t stream) {
    const float* ego       = (const float*)d_in[0];
    const float* entities  = (const float*)d_in[1];
    const int*   n_ent     = (const int*)  d_in[2];
    const float* ego_w     = (const float*)d_in[3];
    const float* ego_b     = (const float*)d_in[4];
    const float* ent_w     = (const float*)d_in[5];
    const float* ent_b     = (const float*)d_in[6];
    const float* ln1g      = (const float*)d_in[7];
    const float* ln1b      = (const float*)d_in[8];
    const float* wq1 = (const float*)d_in[9],  *bq1 = (const float*)d_in[10];
    const float* wk1 = (const float*)d_in[11], *bk1 = (const float*)d_in[12];
    const float* wv1 = (const float*)d_in[13], *bv1 = (const float*)d_in[14];
    const float* wo1 = (const float*)d_in[15], *bo1 = (const float*)d_in[16];
    const float* ln2g      = (const float*)d_in[17];
    const float* ln2b      = (const float*)d_in[18];
    const float* wq2 = (const float*)d_in[19], *bq2 = (const float*)d_in[20];
    const float* wk2 = (const float*)d_in[21], *bk2 = (const float*)d_in[22];
    const float* wv2 = (const float*)d_in[23], *bv2 = (const float*)d_in[24];
    const float* wo2 = (const float*)d_in[25], *bo2 = (const float*)d_in[26];
    const float* fc1_w = (const float*)d_in[27], *fc1_b = (const float*)d_in[28];
    const float* fc2_w = (const float*)d_in[29], *fc2_b = (const float*)d_in[30];
    const float* actor_w = (const float*)d_in[31], *actor_b = (const float*)d_in[32];
    const float* val_w   = (const float*)d_in[33], *val_b   = (const float*)d_in[34];

    __bf16* trunk = (__bf16*)d_ws;                    // [4096, 128] bf16 = 1 MB
    float*  out   = (float*)d_out;                    // logits[4096*4] ++ values[4096]

    policy_attn_kernel<<<NB, 256, 0, stream>>>(
        ego, entities, n_ent, ego_w, ego_b, ent_w, ent_b,
        ln1g, ln1b, wq1, bq1, wk1, bk1, wv1, bv1, wo1, bo1,
        ln2g, ln2b, wq2, bq2, wk2, bk2, wv2, bv2, wo2, bo2, trunk);

    mlp_head_kernel<<<NB / 32, 256, 0, stream>>>(
        trunk, fc1_w, fc1_b, fc2_w, fc2_b, actor_w, actor_b, val_w, val_b, out);
}
